// BrainQuantumLayer_25718264169098
// MI455X (gfx1250) — matrix operand order repaired and checked by reading, not compiled, not run on hardware
//
#include <hip/hip_runtime.h>
#include <math.h>

#define TOKENS  4096
#define IN_DIM  1024
#define NN      2048
#define OUT_DIM 1024
#define TSTEPS  3

typedef __attribute__((ext_vector_type(16))) __bf16 v16bf;
typedef __attribute__((ext_vector_type(2)))  __bf16 v2bf;
typedef __attribute__((ext_vector_type(2)))  float  v2f;
typedef __attribute__((ext_vector_type(8)))  float  v8f;

// native packed f32->bf16 (RNE): selects two-operand v_cvt_pk_bf16_f32
__device__ __forceinline__ unsigned pack_bf16(float x0, float x1) {
    union { v2bf h; unsigned u; } c;
    c.h = __builtin_convertvector((v2f){x0, x1}, v2bf);
    return c.u;
}
__device__ __forceinline__ unsigned short f2bf1(float x) {
    union { __bf16 h; unsigned short u; } c;
    c.h = (__bf16)x;
    return c.u;
}

// gfx1250 hardware tanh (TRANS32 pipe, co-executes with WMMA)
__device__ __forceinline__ float fast_tanh(float x) {
    return __builtin_amdgcn_tanhf(x);
}

union Frag { v16bf v; uint4 q[2]; };

// gfx1250 async global->LDS copy (ASYNCcnt-tracked), 16B per lane.
__device__ __forceinline__ void async_b128_to_lds(unsigned lds_off, const void* gptr) {
    asm volatile("global_load_async_to_lds_b128 %0, %1, off"
                 :: "v"(lds_off), "v"((unsigned long long)(size_t)gptr) : "memory");
}
__device__ __forceinline__ void wait_async0() {
    asm volatile("s_wait_asynccnt 0" ::: "memory");
}

#define EPI_BIAS   0
#define EPI_STORE  1
#define EPI_UPDATE 2

// C[M,N] = A[M,K](fp32, opt tanh) x B[N,K](bf16 row-major) + fused epilogue.
// Block = 128(M) x 256(N), 256 threads = 8 waves in 2(M) x 4(N) grid; wave = 64x64 = 16 WMMA tiles.
// Double-buffered LDS; A piped through registers (needs convert/tanh), B via async-to-LDS.
template<bool A_TANH, int EPI>
__global__ __launch_bounds__(256)
void gemm_wmma(const float* A, const unsigned short* __restrict__ Bg,
               float* Out, int M, int N, int K,
               const float* __restrict__ bias,
               const float* signal, const float* stateOld,
               const float* __restrict__ noise,
               const float* __restrict__ Tvec,
               const float* __restrict__ lamPtr) {
    constexpr int LDSW = 40;                            // padded bf16 row stride (80 B)
    __shared__ __align__(16) unsigned short As[2][128 * LDSW];
    __shared__ __align__(16) unsigned short Bs[2][256 * LDSW];

    const int tid   = threadIdx.x;
    const int lane  = tid & 31;
    const int wave  = tid >> 5;
    const int waveM = wave >> 2;                        // 0..1
    const int waveN = wave & 3;                         // 0..3
    const int blockN = blockIdx.x * 256;
    const int blockM = blockIdx.y * 128;

    const unsigned bsOff[2] = { (unsigned)(size_t)&Bs[0][0], (unsigned)(size_t)&Bs[1][0] };

    v8f acc[4][4];
    #pragma unroll
    for (int i = 0; i < 4; ++i)
        #pragma unroll
        for (int j = 0; j < 4; ++j)
            acc[i][j] = (v8f){0.f,0.f,0.f,0.f,0.f,0.f,0.f,0.f};

    const int aRow = (tid >> 3), aC4 = (tid & 7);       // A: thread -> 4 rows (stride 32), 1 float4 each
    const int nk = K / 32;

    // ---- prologue: prefetch A(k=0) into regs, async B(k=0) -> LDS buf 0 ----
    float4 ar[4];
    #pragma unroll
    for (int it = 0; it < 4; ++it)
        ar[it] = *(const float4*)(A + (size_t)(blockM + aRow + it * 32) * K + aC4 * 4);
    #pragma unroll
    for (int it = 0; it < 4; ++it) {
        int idx = tid + it * 256;                       // 0..1023
        int row = idx >> 2, c = idx & 3;                // 256 rows x 4 b128 per row
        async_b128_to_lds(bsOff[0] + row * 80 + c * 16,
                          Bg + (size_t)(blockN + row) * K + c * 8);
    }

    for (int kk = 0; kk < nk; ++kk) {
        const int p = kk & 1;
        // ---- convert + store prefetched A regs into As[p] ----
        #pragma unroll
        for (int it = 0; it < 4; ++it) {
            float x0 = ar[it].x, x1 = ar[it].y, x2 = ar[it].z, x3 = ar[it].w;
            if (A_TANH) {
                x0 = fast_tanh(x0); x1 = fast_tanh(x1);
                x2 = fast_tanh(x2); x3 = fast_tanh(x3);
            }
            uint2 pk;
            pk.x = pack_bf16(x0, x1);
            pk.y = pack_bf16(x2, x3);
            *(uint2*)(&As[p][(aRow + it * 32) * LDSW + aC4 * 4]) = pk;
        }
        wait_async0();                                  // B[p] fully landed in LDS
        __syncthreads();                                // all waves' A stores + B async visible

        // ---- prefetch next k-chunk (overlaps with WMMAs below) ----
        if (kk + 1 < nk) {
            const int k0n = (kk + 1) * 32;
            #pragma unroll
            for (int it = 0; it < 4; ++it)
                ar[it] = *(const float4*)(A + (size_t)(blockM + aRow + it * 32) * K + k0n + aC4 * 4);
            #pragma unroll
            for (int it = 0; it < 4; ++it) {
                int idx = tid + it * 256;
                int row = idx >> 2, c = idx & 3;
                async_b128_to_lds(bsOff[p ^ 1] + row * 80 + c * 16,
                                  Bg + (size_t)(blockN + row) * K + k0n + c * 8);
            }
        }

        // ---- fragment gathers (ISA A 16x32 swizzle; B per sparse-B-table analogy) ----
        Frag a[4], b[4];
        const int mh = lane & 15, kh = lane >> 4;
        #pragma unroll
        for (int i = 0; i < 4; ++i) {
            int m = waveM * 64 + i * 16 + mh;
            int kb = kh * 8;
            a[i].q[0] = *(const uint4*)(&As[p][m * LDSW + kb]);
            a[i].q[1] = *(const uint4*)(&As[p][m * LDSW + kb + 16]);
        }
        #pragma unroll
        for (int j = 0; j < 4; ++j) {
            int n = waveN * 64 + j * 16 + mh;
            int kb = kh * 8;
            b[j].q[0] = *(const uint4*)(&Bs[p][n * LDSW + kb]);
            b[j].q[1] = *(const uint4*)(&Bs[p][n * LDSW + kb + 16]);
        }
        #pragma unroll
        for (int i = 0; i < 4; ++i)
            #pragma unroll
            for (int j = 0; j < 4; ++j)
                acc[i][j] = __builtin_amdgcn_wmma_f32_16x16x32_bf16(
                    false, a[i].v, false, b[j].v, (short)0, acc[i][j], false, false);
    }

    // ---- epilogue (C/D layout: N = lane&15; M = vgpr + 8*(lane>>4)) ----
    const int mh = lane & 15, half = lane >> 4;
    float lam = 0.f;
    if (EPI == EPI_UPDATE) lam = lamPtr[0];
    #pragma unroll
    for (int i = 0; i < 4; ++i) {
        #pragma unroll
        for (int j = 0; j < 4; ++j) {
            int ng = blockN + waveN * 64 + j * 16 + mh;
            #pragma unroll
            for (int r = 0; r < 8; ++r) {
                int mg = blockM + waveM * 64 + i * 16 + half * 8 + r;
                size_t idx = (size_t)mg * N + ng;
                float v = acc[i][j][r];
                if (EPI == EPI_BIAS) {
                    Out[idx] = v + bias[ng];
                } else if (EPI == EPI_STORE) {
                    Out[idx] = v;
                } else {
                    float s = fast_tanh(stateOld[idx]);
                    Out[idx] = fast_tanh(signal[idx] + lam * (v * s)
                                         + noise[idx] * Tvec[ng] * 0.1f);
                }
            }
        }
    }
}

// ---------------- prep kernels ----------------
__global__ void k_f32_to_bf16(const float* __restrict__ in, unsigned short* __restrict__ out, int n) {
    int i = blockIdx.x * blockDim.x + threadIdx.x;
    if (i < n) out[i] = f2bf1(in[i]);
}

// out_bf[n*NN + k] = w[k*NN + n] * mask[k*NN + n]   (masked transpose, LDS-tiled)
__global__ void k_masked_transpose_bf16(const float* __restrict__ w, const float* __restrict__ mask,
                                        unsigned short* __restrict__ out) {
    __shared__ unsigned short tile[32][33];
    int k0 = blockIdx.y * 32, n0 = blockIdx.x * 32;
    int tx = threadIdx.x & 31, ty = threadIdx.x >> 5;   // 256 thr: ty 0..7
    for (int r = ty; r < 32; r += 8) {
        size_t src = (size_t)(k0 + r) * NN + (n0 + tx);
        tile[r][tx] = f2bf1(w[src] * mask[src]);
    }
    __syncthreads();
    for (int r = ty; r < 32; r += 8)
        out[(size_t)(n0 + r) * NN + (k0 + tx)] = tile[tx][r];
}

__global__ void k_temp(const float* __restrict__ theta, float* __restrict__ Tv) {
    int i = blockIdx.x * blockDim.x + threadIdx.x;
    if (i < NN) Tv[i] = fabsf(sinf(2.0f * theta[i]));
}

// ---------------- launcher ----------------
extern "C" void kernel_launch(void* const* d_in, const int* in_sizes, int n_in,
                              void* d_out, int out_size, void* d_ws, size_t ws_size,
                              hipStream_t stream) {
    const float* x       = (const float*)d_in[0];
    const float* W_in    = (const float*)d_in[1];
    const float* b_in    = (const float*)d_in[2];
    const float* weights = (const float*)d_in[3];
    const float* J       = (const float*)d_in[4];
    const float* theta   = (const float*)d_in[5];
    const float* lam     = (const float*)d_in[6];
    const float* mask    = (const float*)d_in[7];
    const float* noise   = (const float*)d_in[8];
    const float* W_out   = (const float*)d_in[9];
    const float* b_out   = (const float*)d_in[10];

    char* ws = (char*)d_ws;
    float* P0 = (float*)ws;                 ws += (size_t)TOKENS * NN * 4;
    float* P1 = (float*)ws;                 ws += (size_t)TOKENS * NN * 4;
    unsigned short* Win_bf  = (unsigned short*)ws; ws += (size_t)NN * IN_DIM * 2;
    unsigned short* Wout_bf = (unsigned short*)ws; ws += (size_t)OUT_DIM * NN * 2;
    unsigned short* effT    = (unsigned short*)ws; ws += (size_t)NN * NN * 2;
    unsigned short* JmT     = (unsigned short*)ws; ws += (size_t)NN * NN * 2;
    float* Tv = (float*)ws;                 ws += (size_t)NN * 4;

    k_f32_to_bf16<<<(NN * IN_DIM + 255) / 256, 256, 0, stream>>>(W_in, Win_bf, NN * IN_DIM);
    k_f32_to_bf16<<<(OUT_DIM * NN + 255) / 256, 256, 0, stream>>>(W_out, Wout_bf, OUT_DIM * NN);
    k_masked_transpose_bf16<<<dim3(NN / 32, NN / 32), 256, 0, stream>>>(weights, mask, effT);
    k_masked_transpose_bf16<<<dim3(NN / 32, NN / 32), 256, 0, stream>>>(J, mask, JmT);
    k_temp<<<(NN + 255) / 256, 256, 0, stream>>>(theta, Tv);

    dim3 blk(256);
    // state = x @ W_in^T + b_in
    gemm_wmma<false, EPI_BIAS><<<dim3(NN / 256, TOKENS / 128), blk, 0, stream>>>(
        x, Win_bf, P0, TOKENS, NN, IN_DIM, b_in, nullptr, nullptr, nullptr, nullptr, nullptr);

    float* cur = P0; float* oth = P1;
    for (int t = 0; t < TSTEPS; ++t) {
        // signal = state @ eff_w   (into `oth`)
        gemm_wmma<false, EPI_STORE><<<dim3(NN / 256, TOKENS / 128), blk, 0, stream>>>(
            cur, effT, oth, TOKENS, NN, NN, nullptr, nullptr, nullptr, nullptr, nullptr, nullptr);
        // state' = tanh(signal + lam*((tanh(state)@J_m)*tanh(state)) + noise*T*0.1), in-place into `oth`
        gemm_wmma<true, EPI_UPDATE><<<dim3(NN / 256, TOKENS / 128), blk, 0, stream>>>(
            cur, JmT, oth, TOKENS, NN, NN, nullptr, oth, cur,
            noise + (size_t)t * TOKENS * NN, Tv, lam);
        float* tmp = cur; cur = oth; oth = tmp;
    }

    // out = state @ W_out^T + b_out
    gemm_wmma<false, EPI_BIAS><<<dim3(OUT_DIM / 256, TOKENS / 128), blk, 0, stream>>>(
        cur, Wout_bf, (float*)d_out, TOKENS, OUT_DIM, NN, b_out,
        nullptr, nullptr, nullptr, nullptr, nullptr);
}
